// MVB_88089779240976
// MI455X (gfx1250) — matrix operand order repaired and checked by reading, not compiled, not run on hardware
//
#include <hip/hip_runtime.h>

typedef __attribute__((ext_vector_type(16))) _Float16 v16h;
typedef __attribute__((ext_vector_type(8)))  float    v8f;
// Under-aligned vector types: backend emits unaligned ds_load_b128
typedef _Float16 h8v __attribute__((ext_vector_type(8), aligned(2)));
typedef float    f4v __attribute__((ext_vector_type(4), aligned(4)));

#define B_DIM   2
#define C_DIM   128
#define NS_DIM  1024
#define NZ_DIM  256
#define NX_DIM  128
#define LSUB    16
#define MSUB    113            // C - LSUB + 1
#define DLOAD   0.05f
#define WPB     8              // waves (=pixels) per block
#define THREADS (WPB * 32)
#define NPIX    (B_DIM * NZ_DIM * NX_DIM)   // 65536

__device__ __forceinline__ v16h concat8(h8v lo, h8v hi8) {
    v16h r;
    #pragma unroll
    for (int e = 0; e < 8; ++e) { r[e] = lo[e]; r[8 + e] = hi8[e]; }
    return r;
}

__global__ __launch_bounds__(THREADS) void mvb_capon_kernel(
    const float* __restrict__ rf,   // (B, C, NS)
    const float* __restrict__ t0,   // (B,)
    const float* __restrict__ d_tx, // (B, NZ, NX)
    const float* __restrict__ d_rx, // (C, NZ, NX)
    const float* __restrict__ fs_p,
    const float* __restrict__ c0_p,
    float* __restrict__ out)        // (B, NZ, NX)
{
    // Split-precision aligned signals: x = h + e;  e16 stores 2*e (see below).
    // Padded to 160 with zeros so A-fragment reads (index up to 142) stay in-bounds.
    __shared__ _Float16 h16[WPB][160];
    __shared__ _Float16 e16[WPB][160];
    __shared__ float    xf32[WPB][C_DIM];
    __shared__ float    Graw[WPB][LSUB][LSUB];     // unsymmetrized G1
    __shared__ float    As[WPB][LSUB][LSUB + 1];   // augmented [sym(G) | 1]

    const int tid  = threadIdx.x;
    const int wv   = tid >> 5;
    const int lane = tid & 31;
    const int pix  = blockIdx.x * WPB + wv;        // grid sized exactly

    const int b   = pix / (NZ_DIM * NX_DIM);
    const int off = pix - b * (NZ_DIM * NX_DIM);   // z*NX + x

    const float fs    = fs_p[0];
    const float c0inv = 1.0f / c0_p[0];
    const float t0b   = t0[b];
    const float dtx   = d_tx[pix];

    // ---- delay & bilinear interpolate: 4 channels per lane --------------
    #pragma unroll
    for (int q = 0; q < 4; ++q) {
        const int c = lane + 32 * q;
        const float drx = d_rx[c * (NZ_DIM * NX_DIM) + off];
        float pos = fs * ((dtx + drx) * c0inv + t0b);
        pos = fminf(fmaxf(pos, 0.0f), (float)(NS_DIM - 1));
        const int   i0   = (int)pos;               // pos >= 0 -> trunc == floor
        const int   i1   = min(i0 + 1, NS_DIM - 1);
        const float frac = pos - (float)i0;
        const float* sig = rf + (size_t)(b * C_DIM + c) * NS_DIM;
        const float v = sig[i0] * (1.0f - frac) + sig[i1] * frac;
        xf32[wv][c] = v;
        const _Float16 h = (_Float16)v;
        h16[wv][c] = h;
        e16[wv][c] = (_Float16)(2.0f * (v - (float)h));  // 2x residual
    }
    h16[wv][128 + lane] = (_Float16)0.0f;          // zero pad 128..159
    e16[wv][128 + lane] = (_Float16)0.0f;
    __syncthreads();

    // ---- G1 = H^T H + H^T (2E) via WMMA; sym(G1) = compensated S^T S ----
    // S[m,l] = x[m+l], m < 113; K padded 113 -> 128 (4 chunks of 32).
    const int  Mrow = lane & 15;
    const bool hi   = lane >= 16;
    // A operand, k order chosen here: per lane one run of 16 consecutive K
    const int sA = (hi ? 16 : 0) + Mrow;
    // B operand, same k order as A: per lane one run of 16 consecutive K
    const int sB = (hi ? 16 : 0) + Mrow;

    v8f acc = {};
    #pragma unroll
    for (int j = 0; j < 4; ++j) {
        const int base = 32 * j;
        const v16h ah = concat8(*(const h8v*)&h16[wv][base + sA],
                                *(const h8v*)&h16[wv][base + sA + 8]);
        v16h bh, b2;
        if (j < 3 || !hi) {
            // all 16 rows m = base + kbase + 0..15 are valid (m <= 111)
            bh = concat8(*(const h8v*)&h16[wv][base + sB],
                         *(const h8v*)&h16[wv][base + sB + 8]);
            b2 = concat8(*(const h8v*)&e16[wv][base + sB],
                         *(const h8v*)&e16[wv][base + sB + 8]);
        } else {
            // chunk 3, lanes 16-31: rows m = 112..127 -> only m == 112 valid
            bh = (v16h){};
            b2 = (v16h){};
            bh[0] = h16[wv][112 + Mrow];
            b2[0] = e16[wv][112 + Mrow];
        }
        acc = __builtin_amdgcn_wmma_f32_16x16x32_f16(false, ah, false, b2,
                                                     (short)0, acc, false, false);
        acc = __builtin_amdgcn_wmma_f32_16x16x32_f16(false, ah, false, bh,
                                                     (short)0, acc, false, false);
    }

    // C/D layout: VGPR r -> row r (lanes 0-15) / row 8+r (lanes 16-31)
    const int row0 = hi ? 8 : 0;
    #pragma unroll
    for (int r = 0; r < 8; ++r)
        Graw[wv][row0 + r][Mrow] = acc[r];

    // ---- subaperture mean x[l] (lanes 0..15), vectorized ---------------
    float xm = 0.0f;
    if (!hi) {
        f4v s = {0.0f, 0.0f, 0.0f, 0.0f};
        #pragma unroll 4
        for (int m = 0; m < 112; m += 4)
            s += *(const f4v*)&xf32[wv][m + Mrow];
        xm = (s[0] + s[1] + s[2] + s[3] + xf32[wv][112 + Mrow])
             * (1.0f / (float)MSUB);
    }
    __syncthreads();

    // symmetrize: As = 0.5*(G1 + G1^T) = H^T H + H^T E + E^T H
    #pragma unroll
    for (int r = 0; r < 8; ++r)
        As[wv][row0 + r][Mrow] = 0.5f * (acc[r] + Graw[wv][Mrow][row0 + r]);
    if (lane < LSUB)
        As[wv][lane][LSUB] = 1.0f;                 // RHS = ones
    __syncthreads();

    // ---- diagonal loading on scaled system: Ghat = G + DL*tr(G)/16 * I --
    float tr = 0.0f;
    #pragma unroll
    for (int i = 0; i < LSUB; ++i) tr += As[wv][i][i];
    if (lane < LSUB)
        As[wv][lane][lane] += (DLOAD / (float)LSUB) * tr;
    __syncthreads();

    // ---- Gauss-Jordan solve (SPD + loading: no pivoting needed) ---------
    for (int k = 0; k < LSUB; ++k) {
        if (lane < LSUB && lane != k) {
            const float f = As[wv][lane][k] / As[wv][k][k];
            #pragma unroll 1
            for (int j2 = k + 1; j2 <= LSUB; ++j2)
                As[wv][lane][j2] -= f * As[wv][k][j2];
        }
        __syncthreads();
    }

    // u = M * Ghat^{-1} 1  (== R^{-1} 1 of the reference, incl. its scale)
    float u = 0.0f;
    if (lane < LSUB)
        u = (float)MSUB * As[wv][lane][LSUB] / As[wv][lane][lane];

    // ---- y = (u . x) / (sum(u) + 1e-10) --------------------------------
    float num = u * xm;
    float den = u;
    #pragma unroll
    for (int ofs = 8; ofs > 0; ofs >>= 1) {
        num += __shfl_down(num, ofs, 32);
        den += __shfl_down(den, ofs, 32);
    }
    if (lane == 0)
        out[pix] = num / (den + 1e-10f);
}

extern "C" void kernel_launch(void* const* d_in, const int* in_sizes, int n_in,
                              void* d_out, int out_size, void* d_ws, size_t ws_size,
                              hipStream_t stream) {
    // setup_inputs order: rf, t0, d_tx, d_rx, fs, f0, c0, apod
    const float* rf   = (const float*)d_in[0];
    const float* t0   = (const float*)d_in[1];
    const float* d_tx = (const float*)d_in[2];
    const float* d_rx = (const float*)d_in[3];
    const float* fs   = (const float*)d_in[4];
    // d_in[5] (f0) and d_in[7] (apod) are unused by the reference
    const float* c0   = (const float*)d_in[6];

    dim3 grid(NPIX / WPB), block(THREADS);
    mvb_capon_kernel<<<grid, block, 0, stream>>>(rf, t0, d_tx, d_rx, fs, c0,
                                                 (float*)d_out);
    (void)in_sizes; (void)n_in; (void)out_size; (void)d_ws; (void)ws_size;
}
